// DynamicChunking_61297773248714
// MI455X (gfx1250) — matrix operand order repaired and checked by reading, not compiled, not run on hardware
//
#include <hip/hip_runtime.h>
#include <hip/hip_bf16.h>

typedef __attribute__((ext_vector_type(16))) __bf16 v16bf;
typedef __attribute__((ext_vector_type(8)))  __bf16 v8bf;
typedef __attribute__((ext_vector_type(8)))  float  v8f;

constexpr int BB = 8;       // batch
constexpr int L  = 8192;    // sequence
constexpr int D  = 512;     // dim
constexpr int TM = 32;      // rows per tile
constexpr int NBPB = L / TM;            // 256 tiles per batch
constexpr int NTILES = BB * NBPB;       // 2048 tiles total

// ---------------------------------------------------------------------------
// Kernel 0: convert W_q, W_k (fp32, row-major [e][d]) to bf16; zero global sums
// ---------------------------------------------------------------------------
__global__ __launch_bounds__(256) void convert_init_kernel(
    const float* __restrict__ Wq, const float* __restrict__ Wk,
    __bf16* __restrict__ Wqb, __bf16* __restrict__ Wkb,
    float* __restrict__ sums) {
  int idx = blockIdx.x * 256 + threadIdx.x;
  if (idx < D * D) {
    Wqb[idx] = (__bf16)Wq[idx];
    Wkb[idx] = (__bf16)Wk[idx];
  }
  if (idx < 2) sums[idx] = 0.0f;
}

// ---------------------------------------------------------------------------
// Kernel 1: fused bf16-WMMA GEMM (q and k for a 32-row tile, +1 extra k row)
//           -> row norms, cosine vs previous row, p, b_hard, tile count
// ---------------------------------------------------------------------------
__global__ __launch_bounds__(256) void gemm_gate_kernel(
    const float*  __restrict__ x,
    const __bf16* __restrict__ Wq,
    const __bf16* __restrict__ Wk,
    float* __restrict__ p_out,
    float* __restrict__ bh_out,
    int*   __restrict__ blkCnt,
    float* __restrict__ sums) {
  __shared__ __bf16 xs[(TM + 1) * D];   // rows r0-1 .. r0+31 of x (bf16)
  __shared__ __bf16 qs[TM * D];         // raw q rows r0 .. r0+31
  __shared__ __bf16 ks[TM * D];         // raw k rows r0-1 .. r0+30
  __shared__ float  red[3 * TM];        // |q|^2, |k|^2, q.k
  __shared__ float  pv[TM];
  __shared__ int    bv[TM];

  const int bb  = blockIdx.x;
  const int b   = bb / NBPB;
  const int r0  = (bb % NBPB) * TM;
  const int tid = threadIdx.x;
  const float* xb = x + (size_t)b * L * D;

  // ---- stage x tile (fp32 -> bf16), row r0-1 zero-padded at batch start ----
  for (int i = tid; i < (TM + 1) * (D / 4); i += 256) {
    int row = i / (D / 4);
    int col = (i % (D / 4)) * 4;
    int gr  = r0 - 1 + row;
    float4 v = make_float4(0.f, 0.f, 0.f, 0.f);
    if (gr >= 0) v = *(const float4*)(xb + (size_t)gr * D + col);
    __bf16* dst = &xs[row * D + col];
    dst[0] = (__bf16)v.x; dst[1] = (__bf16)v.y;
    dst[2] = (__bf16)v.z; dst[3] = (__bf16)v.w;
  }
  __syncthreads();

  const int wave = tid >> 5;
  const int lane = tid & 31;
  const int lrow = lane & 15;   // row-within-tile (A) / col (B,D)
  const int lhi  = lane >> 4;   // K-half selector

  // ---- two GEMM passes: pass 0 -> q (row offset 1), pass 1 -> k (offset 0)
  for (int pass = 0; pass < 2; ++pass) {
    const __bf16* W  = pass ? Wk : Wq;
    __bf16* dst      = pass ? ks : qs;
    const int roff   = pass ? 0 : 1;

    v8f acc[2][4] = {};   // 2 M-tiles x 4 N-tiles per wave
    for (int kt = 0; kt < D / 32; ++kt) {
      // A fragments (16x32 bf16): lane holds row M=lrow,
      // K = kt*32 + lhi*16 .. +15 contiguous (matches B).
      v16bf afrag[2];
      const int kb = kt * 32 + lhi * 16;
      #pragma unroll
      for (int mt = 0; mt < 2; ++mt) {
        const __bf16* src = &xs[(roff + mt * 16 + lrow) * D + kb];
        v8bf lo = *(const v8bf*)(src);
        v8bf hi = *(const v8bf*)(src + 8);
        afrag[mt] = __builtin_shufflevector(lo, hi,
            0, 1, 2, 3, 4, 5, 6, 7, 8, 9, 10, 11, 12, 13, 14, 15);
      }
      // B fragments (32x16 bf16): lane holds col N=lrow,
      // K = kt*32 + lhi*16 .. +15 contiguous; B[k][n] = W[n][k] (row-major W).
      const int kb2 = kt * 32 + lhi * 16;
      #pragma unroll
      for (int nt = 0; nt < 4; ++nt) {
        const int col = (wave * 4 + nt) * 16 + lrow;
        v16bf bfrag = *(const v16bf*)(W + (size_t)col * D + kb2);
        #pragma unroll
        for (int mt = 0; mt < 2; ++mt) {
          acc[mt][nt] = __builtin_amdgcn_wmma_f32_16x16x32_bf16(
              false, afrag[mt], false, bfrag, (short)0, acc[mt][nt],
              false, false);
        }
      }
    }
    // D layout: VGPR v, lane -> M = v + lhi*8 (+mt*16), N = lrow (+ntile*16)
    #pragma unroll
    for (int mt = 0; mt < 2; ++mt)
      #pragma unroll
      for (int nt = 0; nt < 4; ++nt)
        #pragma unroll
        for (int v = 0; v < 8; ++v) {
          int m = mt * 16 + v + (lhi << 3);
          int n = (wave * 4 + nt) * 16 + lrow;
          dst[m * D + n] = (__bf16)acc[mt][nt][v];
        }
  }
  __syncthreads();

  // ---- row reductions: |q_l|^2, |k_{l-1}|^2, q_l . k_{l-1} ----
  if (tid < 3 * TM) red[tid] = 0.0f;
  __syncthreads();
  {
    const int row = tid & (TM - 1);
    const int seg = tid >> 5;               // 8 segments x 64 elems = 512
    float sq = 0.f, sk = 0.f, sd = 0.f;
    for (int e = seg * 64; e < seg * 64 + 64; ++e) {
      float qvv = (float)qs[row * D + e];
      float kvv = (float)ks[row * D + e];
      sq += qvv * qvv; sk += kvv * kvv; sd += qvv * kvv;
    }
    atomicAdd(&red[row], sq);
    atomicAdd(&red[TM + row], sk);
    atomicAdd(&red[2 * TM + row], sd);
  }
  __syncthreads();

  if (tid < TM) {
    const int gl = r0 + tid;                 // row within batch
    float nq = fmaxf(sqrtf(red[tid]), 1e-12f);
    float nk = fmaxf(sqrtf(red[TM + tid]), 1e-12f);
    float cs = red[2 * TM + tid] / (nq * nk);
    float p  = fminf(fmaxf(0.5f * (1.0f - cs), 0.0f), 1.0f);
    if (gl == 0) p = 1.0f;
    float bh = (p >= 0.5f) ? 1.0f : 0.0f;
    p_out[(size_t)b * L + gl]  = p;
    bh_out[(size_t)b * L + gl] = bh;
    pv[tid] = p; bv[tid] = (int)bh;
  }
  __syncthreads();
  if (tid == 0) {
    float sp = 0.f; int sb = 0;
    for (int i = 0; i < TM; ++i) { sp += pv[i]; sb += bv[i]; }
    blkCnt[bb] = sb;
    atomicAdd(&sums[0], sp);          // sum of p  (G)
    atomicAdd(&sums[1], (float)sb);   // sum of b_hard (F)
  }
}

// ---------------------------------------------------------------------------
// Kernel 2: per-batch exclusive scan of tile counts (trivial serial: 256 vals)
// ---------------------------------------------------------------------------
__global__ void scan_kernel(const int* __restrict__ blkCnt,
                            int* __restrict__ blkOff,
                            int* __restrict__ counts) {
  int b = blockIdx.x;
  if (threadIdx.x == 0) {
    int acc = 0;
    for (int i = 0; i < NBPB; ++i) {
      blkOff[b * NBPB + i] = acc;
      acc += blkCnt[b * NBPB + i];
    }
    counts[b] = acc;
  }
}

// ---------------------------------------------------------------------------
// Kernel 3: zero the compressed output (reference scatters into zeros)
// ---------------------------------------------------------------------------
__global__ __launch_bounds__(256) void zero_kernel(float4* __restrict__ out,
                                                   int n4) {
  int idx = blockIdx.x * 256 + threadIdx.x;
  if (idx < n4) out[idx] = make_float4(0.f, 0.f, 0.f, 0.f);
}

// ---------------------------------------------------------------------------
// Kernel 4: comp_mask, zero p_compressed, ratio loss scalar
// ---------------------------------------------------------------------------
__global__ __launch_bounds__(256) void mask_loss_kernel(
    const int* __restrict__ counts, const float* __restrict__ sums,
    float* __restrict__ mask_out, float* __restrict__ pcomp_out,
    float* __restrict__ loss_out) {
  int idx = blockIdx.x * 256 + threadIdx.x;
  if (idx < BB * L) {
    int b = idx / L, j = idx % L;
    mask_out[idx]  = (j < counts[b]) ? 1.0f : 0.0f;
    pcomp_out[idx] = 0.0f;
  }
  if (idx == 0) {
    const float invBL = 1.0f / (float)(BB * L);
    float G = sums[0] * invBL;
    float F = sums[1] * invBL;
    const float N = 6.0f;   // 1 / TARGET_RATIO
    loss_out[0] = N / (N - 1.0f) *
                  ((N - 1.0f) * F * G + (1.0f - F) * (1.0f - G));
  }
}

// ---------------------------------------------------------------------------
// Kernel 5: scatter — wave32 ballot prefix inside each 32-row tile
// ---------------------------------------------------------------------------
__global__ __launch_bounds__(256) void scatter_kernel(
    const float* __restrict__ x, const float* __restrict__ p_out,
    const int* __restrict__ blkOff,
    float* __restrict__ compressed, float* __restrict__ pcomp) {
  __shared__ int posA[TM];
  const int bb  = blockIdx.x;
  const int b   = bb / NBPB;
  const int r0  = (bb % NBPB) * TM;
  const int tid = threadIdx.x;

  if (tid < TM) {                       // exactly wave 0 (wave32)
    const int gl = r0 + tid;
    float p = p_out[(size_t)b * L + gl];
    int bh = (p >= 0.5f) ? 1 : 0;
    unsigned long long m = __ballot(bh);
    int prefix = __popcll(m & ((1ull << tid) - 1ull));
    int pos = bh ? (blkOff[bb] + prefix) : -1;
    posA[tid] = pos;
    if (pos >= 0) pcomp[(size_t)b * L + pos] = p;
  }
  __syncthreads();

  const float* xb = x + (size_t)b * L * D;
  float* cb = compressed + (size_t)b * L * D;
  for (int r = 0; r < TM; ++r) {
    int pos = posA[r];
    if (pos < 0) continue;
    const float4* src = (const float4*)(xb + (size_t)(r0 + r) * D);
    float4* dst = (float4*)(cb + (size_t)pos * D);
    for (int j = tid; j < D / 4; j += 256) dst[j] = src[j];
  }
}

// ---------------------------------------------------------------------------
extern "C" void kernel_launch(void* const* d_in, const int* in_sizes, int n_in,
                              void* d_out, int out_size, void* d_ws,
                              size_t ws_size, hipStream_t stream) {
  const float* x  = (const float*)d_in[0];
  const float* Wq = (const float*)d_in[1];
  const float* Wk = (const float*)d_in[2];

  // d_out layout (return order, all float):
  // compressed [B*L*D] | mask [B*L] | b_hard [B*L] | p [B*L] | p_comp [B*L] | loss [1]
  float* out = (float*)d_out;
  const size_t oC = 0;
  const size_t oM = (size_t)BB * L * D;
  const size_t oB = oM + (size_t)BB * L;
  const size_t oP = oB + (size_t)BB * L;
  const size_t oPC = oP + (size_t)BB * L;
  const size_t oLoss = oPC + (size_t)BB * L;

  // workspace layout
  char* ws = (char*)d_ws;
  __bf16* Wqb   = (__bf16*)(ws);                       // 512 KB
  __bf16* Wkb   = (__bf16*)(ws + (size_t)D * D * 2);   // 512 KB
  char*   ws2   = ws + (size_t)2 * D * D * 2;
  int*    blkCnt = (int*)(ws2);
  int*    blkOff = (int*)(ws2 + NTILES * 4);
  int*    counts = (int*)(ws2 + 2 * NTILES * 4);
  float*  sums   = (float*)(ws2 + 2 * NTILES * 4 + 64);

  // 0) convert weights to bf16, zero accumulators
  convert_init_kernel<<<(D * D + 255) / 256, 256, 0, stream>>>(Wq, Wk, Wqb,
                                                               Wkb, sums);
  // 1) fused WMMA GEMM + boundary gate
  gemm_gate_kernel<<<NTILES, 256, 0, stream>>>(x, Wqb, Wkb, out + oP, out + oB,
                                               blkCnt, sums);
  // 2) per-batch scan of tile counts
  scan_kernel<<<BB, 32, 0, stream>>>(blkCnt, blkOff, counts);
  // 3) zero compressed output
  {
    int n4 = BB * L * (D / 4);
    zero_kernel<<<(n4 + 255) / 256, 256, 0, stream>>>((float4*)(out + oC), n4);
  }
  // 4) mask, zero p_compressed, ratio loss
  mask_loss_kernel<<<(BB * L + 255) / 256, 256, 0, stream>>>(
      counts, sums, out + oM, out + oPC, out + oLoss);
  // 5) scatter x rows and p into compressed outputs
  scatter_kernel<<<NTILES, 256, 0, stream>>>(x, out + oP, blkOff, out + oC,
                                             out + oPC);
}
